// AttentionalFactorizationMachine_55087250539226
// MI455X (gfx1250) — matrix operand order repaired and checked by reading, not compiled, not run on hardware
//
#include <hip/hip_runtime.h>
#include <math.h>

typedef __attribute__((ext_vector_type(16))) _Float16 v16h;
typedef __attribute__((ext_vector_type(8)))  float    v8f;

#define NF     40
#define ED     64          // EMBED_DIM == HIDDEN == 64
#define NPAIR  780
#define NTM    49           // ceil(780/16)
#define TPB    256
#define NWAVE  8

__global__ __launch_bounds__(TPB)
void afm_kernel(const float* __restrict__ x,
                const float* __restrict__ W1,
                const float* __restrict__ b1,
                const float* __restrict__ W2,
                const float* __restrict__ Wl,
                const float* __restrict__ bl,
                float* __restrict__ out)
{
    __shared__ float xs[NF * ED];          // 10240 B : x[b]
    __shared__ float w1s[ED * ED];         // 16384 B : W1 row-major [h][e]
    __shared__ float b1s[ED];
    __shared__ float w2s[ED];
    __shared__ float wls[ED];
    __shared__ float logits[NPAIR];        // logits, then exp values
    __shared__ unsigned short pr[NPAIR];   // (r<<8)|c
    __shared__ float red[NWAVE];
    __shared__ float red2[ED];
    __shared__ float pooled_part[4][ED];
    __shared__ float sm_max, sm_sum;

    const int b    = blockIdx.x;
    const int tid  = threadIdx.x;
    const int lane = tid & 31;
    const int wave = tid >> 5;

    // ---- stage inputs into LDS ----
    const float* xb = x + (size_t)b * (NF * ED);
    for (int i = tid; i < NF * ED; i += TPB) xs[i]  = xb[i];
    for (int i = tid; i < ED * ED; i += TPB) w1s[i] = W1[i];
    if (tid < ED) { b1s[tid] = b1[tid]; w2s[tid] = W2[tid]; wls[tid] = Wl[tid]; }

    // ---- triangular pair table: p -> (r,c), triu k=1 row-major ----
    for (int p = tid; p < NPAIR; p += TPB) {
        int pp = (NPAIR - 1) - p;
        int k  = (int)((sqrtf(8.0f * (float)pp + 1.0f) - 1.0f) * 0.5f);
        while ((k + 1) * (k + 2) / 2 <= pp) k++;
        while (k * (k + 1) / 2 > pp)       k--;
        int r = NF - 2 - k;
        int S = r * (2 * NF - 1 - r) / 2;       // pairs before row r
        int c = r + 1 + (p - S);
        pr[p] = (unsigned short)((r << 8) | c);
    }
    __syncthreads();

    // ---- build B fragments for W1 (kept in VGPRs for whole kernel) ----
    // B 32x16 in contiguous k order (same as A): lane holds column n=lane&15,
    // half i -> k = i + (lane&16)   (lower lanes K=0..15, upper K=16..31)
    v16h Bf[2][4];
    {
        const int n0   = lane & 15;
        const int koff = lane & 16;
        #pragma unroll
        for (int ks = 0; ks < 2; ++ks)
            #pragma unroll
            for (int nt = 0; nt < 4; ++nt) {
                const int n = nt * 16 + n0;
                #pragma unroll
                for (int i = 0; i < 16; ++i) {
                    const int k = ks * 32 + koff + i;
                    Bf[ks][nt][i] = (_Float16)w1s[n * ED + k];  // W1[n][k]
                }
            }
    }

    // ---- main loop: each wave owns M-tiles of 16 pairs ----
    const int mrow = lane & 15;                 // A row within tile
    const int akoff = (lane & 16) >> 1;         // A-layout lane K offset
    for (int mt = wave; mt < NTM; mt += NWAVE) {
        const int p = mt * 16 + mrow;
        int r = 0, c = 0;
        if (p < NPAIR) { r = pr[p] >> 8; c = pr[p] & 255; }
        const float* xr = xs + r * ED;
        const float* xc = xs + c * ED;

        // A fragments: A 16x32 in contiguous k order (same as B): half i -> k = i + (lane&16)
        v16h A[2];
        #pragma unroll
        for (int ks = 0; ks < 2; ++ks)
            #pragma unroll
            for (int i = 0; i < 16; ++i) {
                const int k = ks * 32 + (lane & 16) + i;
                A[ks][i] = (_Float16)(xr[k] * xc[k]);
            }

        v8f acc[4];
        #pragma unroll
        for (int nt = 0; nt < 4; ++nt) {
            v8f cacc = {0.f,0.f,0.f,0.f,0.f,0.f,0.f,0.f};
            cacc = __builtin_amdgcn_wmma_f32_16x16x32_f16(false, A[0], false, Bf[0][nt],
                                                          (short)0, cacc, false, false);
            cacc = __builtin_amdgcn_wmma_f32_16x16x32_f16(false, A[1], false, Bf[1][nt],
                                                          (short)0, cacc, false, false);
            acc[nt] = cacc;
        }

        // epilogue: s[j] = sum_n relu(h) * W2[n] for row (mt*16 + j + 8*(lane>=16))
        const int n0 = lane & 15;
        float s[8];
        #pragma unroll
        for (int j = 0; j < 8; ++j) {
            float t = 0.f;
            #pragma unroll
            for (int nt = 0; nt < 4; ++nt) {
                const int n = nt * 16 + n0;
                float h = acc[nt][j] + b1s[n];
                h = fmaxf(h, 0.f);
                t += h * w2s[n];
            }
            s[j] = t;
        }
        // reduce across each 16-lane group (columns)
        #pragma unroll
        for (int m = 1; m <= 8; m <<= 1)
            #pragma unroll
            for (int j = 0; j < 8; ++j)
                s[j] += __shfl_xor(s[j], m, 32);

        const int j = lane & 15;
        if (j < 8) {
            const int prow = mt * 16 + j + ((lane & 16) >> 1);
            if (prow < NPAIR) logits[prow] = s[j];
        }
    }
    __syncthreads();

    // ---- softmax over 780 pairs (b2 cancels under softmax) ----
    float lmax = -__builtin_inff();
    for (int p = tid; p < NPAIR; p += TPB) lmax = fmaxf(lmax, logits[p]);
    #pragma unroll
    for (int m = 1; m < 32; m <<= 1) lmax = fmaxf(lmax, __shfl_xor(lmax, m, 32));
    if (lane == 0) red[wave] = lmax;
    __syncthreads();
    if (tid == 0) {
        float m = red[0];
        for (int w = 1; w < NWAVE; ++w) m = fmaxf(m, red[w]);
        sm_max = m;
    }
    __syncthreads();
    const float gmax = sm_max;

    float lsum = 0.f;
    for (int p = tid; p < NPAIR; p += TPB) {
        float e = __expf(logits[p] - gmax);
        logits[p] = e;                 // each p owned by exactly one thread
        lsum += e;
    }
    #pragma unroll
    for (int m = 1; m < 32; m <<= 1) lsum += __shfl_xor(lsum, m, 32);
    if (lane == 0) red[wave] = lsum;
    __syncthreads();
    if (tid == 0) {
        float ssum = 0.f;
        for (int w = 0; w < NWAVE; ++w) ssum += red[w];
        sm_sum = ssum;
    }
    __syncthreads();
    const float inv = 1.0f / sm_sum;

    // ---- pooled[e] = sum_p attn[p] * inner[p][e]; inner recomputed from LDS ----
    {
        const int e     = tid & 63;
        const int chunk = tid >> 6;            // 0..3
        float accp = 0.f;
        for (int p = chunk; p < NPAIR; p += 4) {
            const int r = pr[p] >> 8, c = pr[p] & 255;
            accp += logits[p] * xs[r * ED + e] * xs[c * ED + e];
        }
        pooled_part[chunk][e] = accp;
    }
    __syncthreads();
    if (tid < ED) {
        float pooled = (pooled_part[0][tid] + pooled_part[1][tid] +
                        pooled_part[2][tid] + pooled_part[3][tid]) * inv;
        red2[tid] = pooled * wls[tid];
    }
    __syncthreads();
    if (tid == 0) {
        float o = bl[0];
        for (int i = 0; i < ED; ++i) o += red2[i];
        out[b] = o;
    }
}

extern "C" void kernel_launch(void* const* d_in, const int* in_sizes, int n_in,
                              void* d_out, int out_size, void* d_ws, size_t ws_size,
                              hipStream_t stream) {
    const float* x  = (const float*)d_in[0];
    const float* W1 = (const float*)d_in[1];
    const float* b1 = (const float*)d_in[2];
    const float* W2 = (const float*)d_in[3];
    // d_in[4] = b2: unused (softmax is shift-invariant)
    const float* Wl = (const float*)d_in[5];
    const float* bl = (const float*)d_in[6];
    float* out = (float*)d_out;

    const int batch = in_sizes[0] / (NF * ED);   // 2048
    afm_kernel<<<batch, TPB, 0, stream>>>(x, W1, b1, W2, Wl, bl, out);
}
